// LongformerEncoderLayer_30872224923762
// MI455X (gfx1250) — compile-verified, matrix operand order checked by reading, not run on hardware
//
#include <hip/hip_runtime.h>
#include <hip/hip_bf16.h>

// ---------------- CDNA5 WMMA types/helpers ----------------
typedef __attribute__((ext_vector_type(16))) __bf16 v16bf;
typedef __attribute__((ext_vector_type(8)))  __bf16 bf16x8;
typedef __attribute__((ext_vector_type(8)))  float  v8f;
typedef __attribute__((ext_vector_type(4)))  unsigned int u32x4;
typedef __attribute__((ext_vector_type(8)))  int i32x8;
typedef __attribute__((ext_vector_type(4)))  int i32x4;

// ROCm 7.2 clang-22 declares the 5-arg tensor_load builtin; the
// amdgpu-toolchain (therock-10.0 headers) declares the 6-arg form and is the
// only one shipping the TDM header -- use that to pick the call shape.
#if defined(__has_include)
#if __has_include(<hip/amd_detail/amd_gfx1250_TDM.h>)
#define TDM_6ARG 1
#else
#define TDM_6ARG 0
#endif
#else
#define TDM_6ARG 0
#endif

__device__ __forceinline__ v8f wmma_bf16(v16bf a, v16bf b, v8f c) {
  // D = A(16x32 bf16) x B(32x16 bf16) + C(16x16 f32)
  return __builtin_amdgcn_wmma_f32_16x16x32_bf16(false, a, false, b, (short)0, c,
                                                 false, false);
}

// Load a 16-element bf16 fragment slice for one lane.
// p points at row-base + kb (16B aligned); elements [kb..kb+7] then [kb+16..kb+23].
__device__ __forceinline__ v16bf ld_frag(const __bf16* p) {
  bf16x8 lo = *(const bf16x8*)p;
  bf16x8 hi = *(const bf16x8*)(p + 16);
  v16bf f;
#pragma unroll
  for (int i = 0; i < 8; ++i) { f[i] = lo[i]; f[i + 8] = hi[i]; }
  return f;
}

// Issue one TDM 2D tile load: tile_rows x tile_cols bf16 elements,
// row stride = stride_elems, from gaddr into LDS byte-offset lds_addr.
// D# bitfields per CDNA5 ISA section 8.3/8.4 (group0 128b, group1 256b).
__device__ __forceinline__ void tdm_load_2d(unsigned lds_addr, const void* gaddr,
                                            unsigned tile_rows, unsigned tile_cols,
                                            unsigned stride_elems) {
  unsigned long long ga = (unsigned long long)(size_t)gaddr;
  u32x4 g0;
  g0[0] = 1u;                                        // count=1 (valid), user mode
  g0[1] = lds_addr;                                  // [63:32] lds_addr
  g0[2] = (unsigned)(ga & 0xffffffffu);              // [95:64] global_addr lo
  g0[3] = (unsigned)((ga >> 32) & 0x01ffffffu)       // [120:96] global_addr hi
          | (2u << 30);                              // [127:126] type=2 (image)
  const unsigned td0 = stride_elems;                 // tensor_dim0 (no OOB hit)
  const unsigned td1 = 0x100000u;                    // tensor_dim1 = 1M rows
  i32x8 g1;
  g1[0] = (int)(1u << 16);                           // data_size=1 (2B); wg_mask=0
  g1[1] = (int)((td0 & 0xffffu) << 16);              // [63:48] tensor_dim0 lo16
  g1[2] = (int)((td0 >> 16) | ((td1 & 0xffffu) << 16)); // dim0 hi16 | dim1 lo16
  g1[3] = (int)((td1 >> 16) | (tile_cols << 16));    // dim1 hi16 | tile_dim0
  g1[4] = (int)(tile_rows & 0xffffu);                // tile_dim1 (tile_dim2=0)
  g1[5] = (int)stride_elems;                         // tensor_dim0_stride lo32
  g1[6] = 0;                                         // stride hi16 | dim1_stride lo
  g1[7] = 0;
  i32x4 z4 = {};
#if TDM_6ARG
  i32x8 z8 = {};
  __builtin_amdgcn_tensor_load_to_lds(g0, g1, z4, z4, z8, 0);
#else
  __builtin_amdgcn_tensor_load_to_lds(g0, g1, z4, z4, 0);
#endif
}

#define SEQ 4096
#define DM  768
#define NH  12
#define HD  64
#define WB  128   // band half-width
#define DFF 3072

// ---------------- elementwise kernels ----------------
__global__ __launch_bounds__(256) void cast_bf16_kernel(const float* __restrict__ in,
                                                        __bf16* __restrict__ out, int n) {
  int i = blockIdx.x * 256 + threadIdx.x;
  if (i < n) out[i] = (__bf16)in[i];
}

// in: (K,N) row-major f32 -> out: (N,K) row-major bf16
__global__ __launch_bounds__(256) void transpose_cast_kernel(const float* __restrict__ in,
                                                             __bf16* __restrict__ out,
                                                             int K, int N) {
  int i = blockIdx.x * 256 + threadIdx.x;
  if (i < N * K) {
    int n = i / K, k = i - n * K;
    out[i] = (__bf16)in[(size_t)k * N + n];
  }
}

__global__ __launch_bounds__(256) void add_residual_kernel(const float* __restrict__ src,
                                                           const float* __restrict__ ctx,
                                                           float* __restrict__ xf,
                                                           __bf16* __restrict__ xb, int n) {
  int i = blockIdx.x * 256 + threadIdx.x;
  if (i < n) {
    float v = src[i] + ctx[i];
    xf[i] = v;
    xb[i] = (__bf16)v;
  }
}

// ---------------- generic bf16 WMMA GEMM with TDM staging ----------------
// C(M,N) = act( (A(M,K) @ BT(N,K)^T + bias) * scale ) [+ addin]
// 128x128 block tile, 256 threads (8 waves), wave tile 32x64, K step 32.
// Tiles are DMA'd into double-buffered LDS by the Tensor Data Mover (wave 0
// issues, TENSORcnt + barrier publishes), overlapping DMA of step s+1 with
// the WMMAs of step s.
__global__ __launch_bounds__(256) void gemm_bf16_kernel(
    const __bf16* __restrict__ A, const __bf16* __restrict__ BT,
    const float* __restrict__ bias, float scale, int relu,
    const float* __restrict__ addin, float* __restrict__ outf,
    __bf16* __restrict__ outb, int M, int N, int K) {
  __shared__ __bf16 Al[2][128 * 32];
  __shared__ __bf16 Bl[2][128 * 32];

  const int tid  = threadIdx.x;
  const int lane = tid & 31;
  const int wave = tid >> 5;
  const int wm = wave >> 1;          // 0..3
  const int wn = wave & 1;           // 0..1
  const int m0 = blockIdx.y * 128;
  const int n0 = blockIdx.x * 128;
  const int half = lane >> 4;
  const int lr   = lane & 15;
  const int kb   = half * 8;

  v8f acc[2][4] = {};

  const int nsteps = K / 32;
  if (wave == 0) {
    tdm_load_2d((unsigned)(size_t)&Al[0][0], &A[(size_t)m0 * K], 128, 32, K);
    tdm_load_2d((unsigned)(size_t)&Bl[0][0], &BT[(size_t)n0 * K], 128, 32, K);
  }

  for (int s = 0; s < nsteps; ++s) {
    const int p = s & 1;
    if (wave == 0) __builtin_amdgcn_s_wait_tensorcnt(0);
    __syncthreads();     // buf[p] ready for all waves; buf[p^1] free to refill
    if (wave == 0 && s + 1 < nsteps) {
      const int k1 = (s + 1) * 32;
      tdm_load_2d((unsigned)(size_t)&Al[p ^ 1][0], &A[(size_t)m0 * K + k1], 128, 32, K);
      tdm_load_2d((unsigned)(size_t)&Bl[p ^ 1][0], &BT[(size_t)n0 * K + k1], 128, 32, K);
    }

    v16bf af[2], bfr[4];
#pragma unroll
    for (int fm = 0; fm < 2; ++fm)
      af[fm] = ld_frag(&Al[p][(wm * 32 + fm * 16 + lr) * 32 + kb]);
#pragma unroll
    for (int fn = 0; fn < 4; ++fn)
      bfr[fn] = ld_frag(&Bl[p][(wn * 64 + fn * 16 + lr) * 32 + kb]);
#pragma unroll
    for (int fm = 0; fm < 2; ++fm)
#pragma unroll
      for (int fn = 0; fn < 4; ++fn)
        acc[fm][fn] = wmma_bf16(af[fm], bfr[fn], acc[fm][fn]);
  }

#pragma unroll
  for (int fm = 0; fm < 2; ++fm)
#pragma unroll
    for (int fn = 0; fn < 4; ++fn)
#pragma unroll
      for (int r = 0; r < 8; ++r) {
        int gm = m0 + wm * 32 + fm * 16 + r + 8 * half;   // C/D: row = r + 8*half
        int gn = n0 + wn * 64 + fn * 16 + lr;             //       col = lane&15
        float v = (acc[fm][fn][r] + bias[gn]) * scale;
        if (relu) v = fmaxf(v, 0.f);
        size_t o = (size_t)gm * N + gn;
        if (addin) v += addin[o];
        if (outf) outf[o] = v;
        if (outb) outb[o] = (__bf16)v;
      }
}

// ---------------- banded attention ----------------
// One block per (chunk, head, batch); 4 waves, each does two 16-row strips.
// Dynamic LDS layout (bytes):
//   [0,      49152)  K-band  [384][64] bf16   (natural layout == B operand of QK^T)
//   [49152,  98304)  V-bandT [64][384] bf16   (B operand of P.V)
//   [98304, 196608)  per-wave score strip [16][384] f32  (4 waves)
//   [196608,245760)  per-wave prob  strip [16][384] bf16 (4 waves)
__global__ __launch_bounds__(128) void band_attn_kernel(
    const __bf16* __restrict__ Qb, const __bf16* __restrict__ Kb,
    const __bf16* __restrict__ Vb, float* __restrict__ ctx,
    float* __restrict__ attw) {
  extern __shared__ char smem[];
  __bf16* Kl = (__bf16*)smem;
  __bf16* VT = (__bf16*)(smem + 49152);
  float*  Sb = (float*)(smem + 98304);
  __bf16* Pb = (__bf16*)(smem + 196608);

  const int chunk = blockIdx.x;   // 0..31
  const int h     = blockIdx.y;   // 0..11
  const int b     = blockIdx.z;   // 0..1
  const int tid   = threadIdx.x;
  const int lane  = tid & 31;
  const int wave  = tid >> 5;
  const int half  = lane >> 4;
  const int lr    = lane & 15;
  const int kb    = half * 8;

  const size_t headoff = (size_t)h * HD;
  const int base = chunk * WB - WB;      // sequence position of band column j=0

  // stage K band (zero-padded outside [0,SEQ))
  for (int c = tid; c < 384 * 8; c += 128) {
    int j = c >> 3, cc = (c & 7) * 8;
    int pos = base + j;
    uint4 val = make_uint4(0u, 0u, 0u, 0u);
    if (pos >= 0 && pos < SEQ)
      val = *(const uint4*)&Kb[((size_t)(b * SEQ + pos)) * DM + headoff + cc];
    *(uint4*)&Kl[j * 64 + cc] = val;
  }
  // stage V band transposed: VT[d][j]
  for (int e = tid; e < 384 * 64; e += 128) {
    int j = e >> 6, d = e & 63;
    int pos = base + j;
    __bf16 v = (__bf16)0.f;
    if (pos >= 0 && pos < SEQ)
      v = Vb[((size_t)(b * SEQ + pos)) * DM + headoff + d];
    VT[d * 384 + j] = v;
  }
  __syncthreads();

  float*  S = Sb + wave * (16 * 384);
  __bf16* P = Pb + wave * (16 * 384);

  for (int ss = 0; ss < 2; ++ss) {
    const int t0 = wave * 32 + ss * 16;     // strip base (within-chunk row)

    // A fragments of Q for the two K-steps over HD=64 (kept in registers)
    v16bf aq[2];
#pragma unroll
    for (int kk = 0; kk < 2; ++kk)
      aq[kk] = ld_frag(&Qb[((size_t)(b * SEQ + chunk * WB + t0 + lr)) * DM +
                           headoff + kk * 32 + kb]);

    // scores: 24 column tiles of 16 over the 384-wide band
    for (int jf = 0; jf < 24; ++jf) {
      v8f d = {};
#pragma unroll
      for (int kk = 0; kk < 2; ++kk) {
        v16bf bk = ld_frag(&Kl[(jf * 16 + lr) * 64 + kk * 32 + kb]);
        d = wmma_bf16(aq[kk], bk, d);
      }
#pragma unroll
      for (int r = 0; r < 8; ++r)
        S[(r + 8 * half) * 384 + jf * 16 + lr] = d[r];
    }

    // masked softmax: 2 lanes per row
    const int row = lane >> 1;
    const int hh  = lane & 1;
    const int tl  = t0 + row;               // within-chunk row 0..127
    float mx = -3.0e38f;
    for (int j = hh; j < 384; j += 2) {
      int pos = base + j;
      bool valid = (j >= tl) && (j <= tl + 2 * WB) && (pos >= 0) && (pos < SEQ);
      if (valid) mx = fmaxf(mx, S[row * 384 + j]);
    }
    mx = fmaxf(mx, __shfl_xor(mx, 1, 32));
    float sum = 0.f;
    for (int j = hh; j < 384; j += 2) {
      int pos = base + j;
      bool valid = (j >= tl) && (j <= tl + 2 * WB) && (pos >= 0) && (pos < SEQ);
      float p = valid ? __expf(S[row * 384 + j] - mx) : 0.f;
      S[row * 384 + j] = p;
      sum += p;
    }
    sum += __shfl_xor(sum, 1, 32);
    const float inv = 1.f / sum;
    for (int j = hh; j < 384; j += 2) {
      float p = S[row * 384 + j] * inv;
      S[row * 384 + j] = p;
      P[row * 384 + j] = (__bf16)p;
    }

    // emit attention-band probabilities: columns [tl, tl+256]
    {
      int tg = chunk * WB + tl;
      float* aw = attw + (((size_t)(b * NH + h) * SEQ) + tg) * (2 * WB + 1);
      for (int c = hh; c < 2 * WB + 1; c += 2) aw[c] = S[row * 384 + tl + c];
    }

    // context: P(16x384) @ Vband(384x64)
    v8f accv[4] = {};
    for (int ks = 0; ks < 12; ++ks) {
      v16bf ap = ld_frag(&P[lr * 384 + ks * 32 + kb]);
#pragma unroll
      for (int fn = 0; fn < 4; ++fn) {
        v16bf bv = ld_frag(&VT[(fn * 16 + lr) * 384 + ks * 32 + kb]);
        accv[fn] = wmma_bf16(ap, bv, accv[fn]);
      }
    }
#pragma unroll
    for (int fn = 0; fn < 4; ++fn)
#pragma unroll
      for (int r = 0; r < 8; ++r) {
        int tm = t0 + r + 8 * half;
        ctx[((size_t)(b * SEQ + chunk * WB + tm)) * DM + headoff + fn * 16 + lr] =
            accv[fn][r];
      }
  }
}

// ---------------- LayerNorm (row = 768) ----------------
__global__ __launch_bounds__(256) void layernorm_kernel(const float* __restrict__ y,
                                                        const float* __restrict__ g,
                                                        const float* __restrict__ beta,
                                                        float* __restrict__ out) {
  __shared__ float red[256];
  const int row = blockIdx.x;
  const float* yr = y + (size_t)row * DM;
  float v[3];
  float s = 0.f;
#pragma unroll
  for (int i = 0; i < 3; ++i) { v[i] = yr[threadIdx.x + 256 * i]; s += v[i]; }
  red[threadIdx.x] = s;
  __syncthreads();
  for (int st = 128; st > 0; st >>= 1) {
    if (threadIdx.x < st) red[threadIdx.x] += red[threadIdx.x + st];
    __syncthreads();
  }
  const float mu = red[0] * (1.f / DM);
  __syncthreads();
  float s2 = 0.f;
#pragma unroll
  for (int i = 0; i < 3; ++i) { float d = v[i] - mu; s2 += d * d; }
  red[threadIdx.x] = s2;
  __syncthreads();
  for (int st = 128; st > 0; st >>= 1) {
    if (threadIdx.x < st) red[threadIdx.x] += red[threadIdx.x + st];
    __syncthreads();
  }
  const float inv = rsqrtf(red[0] * (1.f / DM) + 1e-6f);
  float* o = out + (size_t)row * DM;
#pragma unroll
  for (int i = 0; i < 3; ++i) {
    int c = threadIdx.x + 256 * i;
    o[c] = g[c] * (v[i] - mu) * inv + beta[c];
  }
}

// ---------------- host orchestration ----------------
extern "C" void kernel_launch(void* const* d_in, const int* in_sizes, int n_in,
                              void* d_out, int out_size, void* d_ws, size_t ws_size,
                              hipStream_t stream) {
  const float* src  = (const float*)d_in[0];
  const float* Wq   = (const float*)d_in[1];
  const float* bq   = (const float*)d_in[2];
  const float* Wk   = (const float*)d_in[3];
  const float* bk   = (const float*)d_in[4];
  const float* Wv   = (const float*)d_in[5];
  const float* bv   = (const float*)d_in[6];
  const float* W1   = (const float*)d_in[7];
  const float* b1   = (const float*)d_in[8];
  const float* W2   = (const float*)d_in[9];
  const float* b2   = (const float*)d_in[10];
  const float* g2   = (const float*)d_in[11];
  const float* bt2  = (const float*)d_in[12];

  const int B = 2;
  const size_t MROWS = (size_t)B * SEQ;            // 8192
  const size_t MD = MROWS * DM;                    // 6,291,456

  // workspace carve-up
  char* ws = (char*)d_ws;
  __bf16* s_bf = (__bf16*)ws;                ws += MD * 2;                 // src bf16
  __bf16* WqT  = (__bf16*)ws;                ws += (size_t)DM * DM * 2;
  __bf16* WkT  = (__bf16*)ws;                ws += (size_t)DM * DM * 2;
  __bf16* WvT  = (__bf16*)ws;                ws += (size_t)DM * DM * 2;
  __bf16* W1T  = (__bf16*)ws;                ws += (size_t)DFF * DM * 2;   // (3072,768)
  __bf16* W2T  = (__bf16*)ws;                ws += (size_t)DM * DFF * 2;   // (768,3072)
  __bf16* Qbf  = (__bf16*)ws;                ws += MD * 2;
  __bf16* Kbf  = (__bf16*)ws;                ws += MD * 2;
  __bf16* Vbf  = (__bf16*)ws;                ws += MD * 2;
  float*  ctx  = (float*)ws;                 ws += MD * 4;
  float*  xf   = (float*)ws;                 ws += MD * 4;
  __bf16* xb   = (__bf16*)ws;                ws += MD * 2;
  __bf16* h1b  = (__bf16*)ws;                ws += MROWS * DFF * 2;
  float*  yf   = (float*)ws;                 ws += MD * 4;

  float* outF  = (float*)d_out;              // (B,S,D)
  float* attw  = outF + MD;                  // (B,H,S,257)

  // 1) casts / weight transposes
  cast_bf16_kernel<<<(int)((MD + 255) / 256), 256, 0, stream>>>(src, s_bf, (int)MD);
  transpose_cast_kernel<<<(DM * DM + 255) / 256, 256, 0, stream>>>(Wq, WqT, DM, DM);
  transpose_cast_kernel<<<(DM * DM + 255) / 256, 256, 0, stream>>>(Wk, WkT, DM, DM);
  transpose_cast_kernel<<<(DM * DM + 255) / 256, 256, 0, stream>>>(Wv, WvT, DM, DM);
  transpose_cast_kernel<<<(DM * DFF + 255) / 256, 256, 0, stream>>>(W1, W1T, DM, DFF);
  transpose_cast_kernel<<<(DFF * DM + 255) / 256, 256, 0, stream>>>(W2, W2T, DFF, DM);

  // 2) QKV projections (Q folds the 1/sqrt(HD) scale)
  dim3 gQKV(DM / 128, (unsigned)(MROWS / 128));
  gemm_bf16_kernel<<<gQKV, 256, 0, stream>>>(s_bf, WqT, bq, 0.125f, 0, nullptr,
                                             nullptr, Qbf, (int)MROWS, DM, DM);
  gemm_bf16_kernel<<<gQKV, 256, 0, stream>>>(s_bf, WkT, bk, 1.0f, 0, nullptr,
                                             nullptr, Kbf, (int)MROWS, DM, DM);
  gemm_bf16_kernel<<<gQKV, 256, 0, stream>>>(s_bf, WvT, bv, 1.0f, 0, nullptr,
                                             nullptr, Vbf, (int)MROWS, DM, DM);

  // 3) banded attention (245760 B dynamic LDS on the 320KB WGP)
  (void)hipFuncSetAttribute((const void*)band_attn_kernel,
                            hipFuncAttributeMaxDynamicSharedMemorySize, 245760);
  band_attn_kernel<<<dim3(SEQ / WB, NH, B), 128, 245760, stream>>>(Qbf, Kbf, Vbf,
                                                                   ctx, attw);

  // 4) x = src + attn_out
  add_residual_kernel<<<(int)((MD + 255) / 256), 256, 0, stream>>>(src, ctx, xf, xb,
                                                                   (int)MD);

  // 5) FFN
  dim3 gF1(DFF / 128, (unsigned)(MROWS / 128));
  gemm_bf16_kernel<<<gF1, 256, 0, stream>>>(xb, W1T, b1, 1.0f, 1, nullptr,
                                            nullptr, h1b, (int)MROWS, DFF, DM);
  dim3 gF2(DM / 128, (unsigned)(MROWS / 128));
  gemm_bf16_kernel<<<gF2, 256, 0, stream>>>(h1b, W2T, b2, 1.0f, 0, xf,
                                            yf, nullptr, (int)MROWS, DM, DFF);

  // 6) LayerNorm -> d_out
  layernorm_kernel<<<(int)MROWS, 256, 0, stream>>>(yf, g2, bt2, outF);
}
